// HyperbolicCNNLayer_33603824124389
// MI455X (gfx1250) — matrix operand order repaired and checked by reading, not compiled, not run on hardware
//
#include <hip/hip_runtime.h>
#include <cmath>

// ---------------------------------------------------------------------------
// Hyperbolic CNN layer for MI455X (gfx1250).
//   Pass 1: split fp32 weights into bf16 hi/lo, pre-transposed per-tile.
//   Pass 2: split fp32 x into bf16 hi/lo flat arrays.
//   Pass 3: implicit-GEMM conv on v_wmma_f32_16x16x32_bf16 (3-WMMA fp32
//           emulation), ALL LDS staging done by the Tensor Data Mover
//           (tensor_load_to_lds + s_wait_tensorcnt), fused window-norm
//           scale, tanh exp-map, bias + ReLU.
// Shapes: x[32,1024,256] f32, kernel[9,256,512] f32, bias[512] f32
// Out:    [32,1016,512] f32
// ---------------------------------------------------------------------------

typedef __attribute__((ext_vector_type(16))) __bf16 v16bf;
typedef __attribute__((ext_vector_type(8)))  float  v8f;
typedef __attribute__((ext_vector_type(4)))  unsigned int u32x4;
typedef __attribute__((ext_vector_type(8)))  int    i32x8;
typedef __attribute__((ext_vector_type(4)))  int    i32x4;

#define HB_B     32
#define HB_L     1024
#define HB_CIN   256
#define HB_K     9
#define HB_F     512
#define HB_LOUT  1016

#define TILE_M   64                       // lo positions per workgroup
#define TILE_N   128                      // filters per workgroup
#define SLAB_R   (TILE_M + HB_K - 1)      // 72 x-rows staged
#define KSTEPS   ((HB_K * HB_CIN) / 32)   // 72 K-steps of 32
#define LO_TILES 16
#define F_TILES  (HB_F / TILE_N)          // 4

#define HB_EPS   1e-7f
#define HB_LAMH  0.99999990f              // 1/(1+eps)

// workspace layout (elements of bf16)
#define WT_TILE_ELEMS 8192                          // [hi 4096 | lo 4096] per (ft,s)
#define WT_ELEMS  (F_TILES * KSTEPS * WT_TILE_ELEMS)   // 2,359,296
#define X_ELEMS   (HB_B * HB_L * HB_CIN)               // 8,388,608

// LDS carve (elements)
#define A_ELEMS  (SLAB_R * HB_CIN)        // 18432 per hi/lo array
#define SMEM_BYTES ((2 * A_ELEMS + 2 * WT_TILE_ELEMS) * 2 + (SLAB_R + TILE_M) * 4)

// ---------------------------------------------------------------------------
__device__ __forceinline__ void split_bf16(float v, __bf16& h, __bf16& l) {
    h = (__bf16)v;
    l = (__bf16)(v - (float)h);
}

__device__ __forceinline__ v16bf ld_frag_2seg(const __bf16* p0, const __bf16* p1) {
    union { v16bf v; u32x4 q[2]; } u;
    u.q[0] = *(const u32x4*)p0;
    u.q[1] = *(const u32x4*)p1;
    return u.v;
}

__device__ __forceinline__ unsigned int lds_offset(const void* p) {
    // flat->LDS mapping keeps the wave-relative byte offset in addr[31:0]
    return (unsigned int)(uintptr_t)p;
}

// 1-D TDM copy: tile_elems bf16 from global -> LDS; reads beyond tensor_elems
// return zero (OOB clamp for the tail slab).  D# per CDNA5 ISA ch.8.
__device__ __forceinline__ void tdm_load_1d(unsigned int lds_byte, const void* gptr,
                                            unsigned int tile_elems,
                                            unsigned int tensor_elems) {
    unsigned long long ga = (unsigned long long)(uintptr_t)gptr;
    u32x4 g0 = { 1u,                                   // count=1, user descriptor
                 lds_byte,                             // lds_addr
                 (unsigned int)ga,                     // global_addr[31:0]
                 (unsigned int)((ga >> 32) & 0x01FFFFFFu) | (2u << 30) }; // type=2
    i32x8 g1 = { (int)(1u << 16),                                  // data_size=2B
                 (int)((tensor_elems & 0xFFFFu) << 16),            // tensor_dim0 lo
                 (int)(((tensor_elems >> 16) & 0xFFFFu) | (1u << 16)), // hi | dim1=1
                 (int)((tile_elems & 0xFFFFu) << 16),              // tile_dim0
                 0,
                 (int)tensor_elems,                                // dim0_stride lo
                 0, 0 };
    i32x4 z4 = { 0, 0, 0, 0 };
#if defined(__clang_major__) && (__clang_major__ >= 23)
    i32x8 z8 = { 0, 0, 0, 0, 0, 0, 0, 0 };
    __builtin_amdgcn_tensor_load_to_lds(g0, g1, z4, z4, z8, 0);
#else
    __builtin_amdgcn_tensor_load_to_lds(g0, g1, z4, z4, 0);
#endif
}

// ---------------------------------------------------------------------------
// Pass 1: weights f32 [2304][512] -> bf16 hi/lo tiles [ft][s][128 n][32 k]
__global__ __launch_bounds__(256)
void prep_weights(const float* __restrict__ w, __bf16* __restrict__ wt) {
    int e = blockIdx.x * 256 + threadIdx.x;
    if (e >= (HB_K * HB_CIN) * HB_F) return;
    int kc = e >> 9, f = e & 511;
    __bf16 h, l;
    split_bf16(w[e], h, l);
    int ft = f >> 7, n = f & 127, s = kc >> 5, k = kc & 31;
    size_t base = (size_t)(ft * KSTEPS + s) * WT_TILE_ELEMS + n * 32 + k;
    wt[base] = h;
    wt[base + 4096] = l;
}

// Pass 2: x f32 -> flat bf16 hi/lo arrays
__global__ __launch_bounds__(256)
void prep_x(const float* __restrict__ x, __bf16* __restrict__ xhi,
            __bf16* __restrict__ xlo) {
    size_t e = (size_t)blockIdx.x * 256 + threadIdx.x;
    if (e >= (size_t)X_ELEMS) return;
    __bf16 h, l;
    split_bf16(x[e], h, l);
    xhi[e] = h;
    xlo[e] = l;
}

// ---------------------------------------------------------------------------
// Pass 3: main WMMA kernel, TDM-staged
__global__ __launch_bounds__(256, 2)
void hyp_conv_wmma_tdm(const __bf16* __restrict__ xhi,
                       const __bf16* __restrict__ xlo,
                       const __bf16* __restrict__ wt,
                       const float*  __restrict__ bias,
                       float* __restrict__ out) {
    extern __shared__ char smem[];
    __bf16* Ahi = (__bf16*)smem;                  // [72][256]
    __bf16* Alo = Ahi + A_ELEMS;
    __bf16* Bt  = Alo + A_ELEMS;                  // [2][hi 4096 | lo 4096]
    float*  rowsq  = (float*)(Bt + 2 * WT_TILE_ELEMS);  // [72]
    float*  wscale = rowsq + SLAB_R;                    // [64]

    const int tid  = threadIdx.x;
    const int lane = tid & 31;
    const int wave = tid >> 5;
    const int wm   = wave & 1;            // 2 waves along M
    const int wn   = wave >> 1;           // 4 waves along N

    const int b   = blockIdx.x >> 4;
    const int lo0 = (blockIdx.x & 15) * TILE_M;
    const int ft  = blockIdx.y;
    const int fb  = ft * TILE_N;

    // ---- TDM: stage x slab (hi+lo) and first weight tile -------------------
    const unsigned int slab_start = (unsigned int)(b * HB_L + lo0) * HB_CIN;
    const unsigned int slab_rem   = (unsigned int)X_ELEMS - slab_start; // OOB clamp
    if (wave == 0) {
        tdm_load_1d(lds_offset(Ahi), xhi + slab_start, A_ELEMS, slab_rem);
        tdm_load_1d(lds_offset(Alo), xlo + slab_start, A_ELEMS, slab_rem);
        tdm_load_1d(lds_offset(Bt),  wt + (size_t)(ft * KSTEPS) * WT_TILE_ELEMS,
                    WT_TILE_ELEMS, WT_TILE_ELEMS);
        __builtin_amdgcn_s_wait_tensorcnt(0);
    }
    __syncthreads();

    // ---- per-row sum of squares from staged slab (shuffle reduce) ----------
    for (int r = wave * 9; r < wave * 9 + 9; ++r) {
        float s = 0.0f;
        const __bf16* ph = Ahi + r * HB_CIN + lane * 8;
        const __bf16* pl = Alo + r * HB_CIN + lane * 8;
        #pragma unroll
        for (int j = 0; j < 8; ++j) {
            float v = (float)ph[j] + (float)pl[j];
            s += v * v;
        }
        #pragma unroll
        for (int off = 16; off; off >>= 1) s += __shfl_xor(s, off, 32);
        if (lane == 0) rowsq[r] = s;
    }
    __syncthreads();
    if (tid < TILE_M) {
        float ws = 0.0f;
        #pragma unroll
        for (int k = 0; k < HB_K; ++k) ws += rowsq[tid + k];
        float wnorm = sqrtf(fmaxf(ws, 0.0f));
        wscale[tid] = (wnorm < 1.0f) ? 1.0f : 1.0f / (wnorm + HB_EPS);
    }

    // ---- main K loop: TDM double-buffered weight tiles ---------------------
    v8f acc[2][2];
    #pragma unroll
    for (int mi = 0; mi < 2; ++mi)
        #pragma unroll
        for (int ni = 0; ni < 2; ++ni)
            #pragma unroll
            for (int r = 0; r < 8; ++r) acc[mi][ni][r] = 0.0f;

    const int m    = lane & 15;
    const int half = lane >> 4;

    for (int s = 0; s < KSTEPS; ++s) {
        const int cur = s & 1;
        if (wave == 0 && s + 1 < KSTEPS) {
            tdm_load_1d(lds_offset(Bt + (WT_TILE_ELEMS << (cur ^ 1 ? 0 : 0)) * (cur ^ 1)),
                        wt + (size_t)(ft * KSTEPS + s + 1) * WT_TILE_ELEMS,
                        WT_TILE_ELEMS, WT_TILE_ELEMS);
        }

        const int tap = s >> 3;
        const int cb  = (s & 7) << 5;
        const __bf16* Bhi = Bt + cur * WT_TILE_ELEMS;
        const __bf16* Blo = Bhi + 4096;

        v16bf ahi[2], alo[2], bhi[2], blo[2];
        #pragma unroll
        for (int mi = 0; mi < 2; ++mi) {
            const int row = wm * 32 + mi * 16 + m + tap;
            const __bf16* p = Ahi + row * HB_CIN + cb + half * 8;
            ahi[mi] = ld_frag_2seg(p, p + 16);
            const __bf16* q = Alo + row * HB_CIN + cb + half * 8;
            alo[mi] = ld_frag_2seg(q, q + 16);
        }
        #pragma unroll
        for (int ni = 0; ni < 2; ++ni) {
            const int col = wn * 32 + ni * 16 + m;
            const __bf16* p = Bhi + col * 32 + half * 8;
            bhi[ni] = ld_frag_2seg(p, p + 16);
            const __bf16* q = Blo + col * 32 + half * 8;
            blo[ni] = ld_frag_2seg(q, q + 16);
        }

        // fp32-via-bf16 split: hi*hi + hi*lo + lo*hi
        #pragma unroll
        for (int mi = 0; mi < 2; ++mi)
            #pragma unroll
            for (int ni = 0; ni < 2; ++ni) {
                acc[mi][ni] = __builtin_amdgcn_wmma_f32_16x16x32_bf16(
                    false, ahi[mi], false, bhi[ni], (short)0, acc[mi][ni], false, false);
                acc[mi][ni] = __builtin_amdgcn_wmma_f32_16x16x32_bf16(
                    false, ahi[mi], false, blo[ni], (short)0, acc[mi][ni], false, false);
                acc[mi][ni] = __builtin_amdgcn_wmma_f32_16x16x32_bf16(
                    false, alo[mi], false, bhi[ni], (short)0, acc[mi][ni], false, false);
            }

        if (wave == 0) __builtin_amdgcn_s_wait_tensorcnt(0);
        __syncthreads();
    }

    // ---- epilogue: scale, tanh exp-map (exp-based), bias, ReLU -------------
    #pragma unroll
    for (int mi = 0; mi < 2; ++mi)
        #pragma unroll
        for (int ni = 0; ni < 2; ++ni) {
            #pragma unroll
            for (int r = 0; r < 8; ++r) {
                const int ml = wm * 32 + mi * 16 + half * 8 + r;
                const int fl = wn * 32 + ni * 16 + m;
                const int lo = lo0 + ml;
                if (lo < HB_LOUT) {
                    float v  = acc[mi][ni][r] * wscale[ml];
                    float vn = fmaxf(fabsf(v), HB_EPS);
                    // tanh(z) = 1 - 2/(exp(2z)+1), z >= 0
                    float t  = 1.0f - 2.0f / (__expf(2.0f * vn * HB_LAMH) + 1.0f);
                    float y  = t * (v / vn);
                    float o  = y + bias[fb + fl];
                    out[((size_t)b * HB_LOUT + lo) * HB_F + fl] = fmaxf(o, 0.0f);
                }
            }
        }
}

// ---------------------------------------------------------------------------
extern "C" void kernel_launch(void* const* d_in, const int* in_sizes, int n_in,
                              void* d_out, int out_size, void* d_ws, size_t ws_size,
                              hipStream_t stream) {
    (void)in_sizes; (void)n_in; (void)out_size; (void)ws_size;
    const float* x    = (const float*)d_in[0];
    const float* kern = (const float*)d_in[1];
    const float* bias = (const float*)d_in[2];
    float* out = (float*)d_out;

    __bf16* wt  = (__bf16*)d_ws;            // 2,359,296 elems (4.7 MB)
    __bf16* xhi = wt + WT_ELEMS;            // 8,388,608 elems (16.8 MB)
    __bf16* xlo = xhi + X_ELEMS;            // 8,388,608 elems (16.8 MB)

    prep_weights<<<dim3(((HB_K * HB_CIN) * HB_F + 255) / 256), dim3(256), 0, stream>>>(
        kern, wt);
    prep_x<<<dim3((X_ELEMS + 255) / 256), dim3(256), 0, stream>>>(x, xhi, xlo);

    hipFuncSetAttribute((const void*)hyp_conv_wmma_tdm,
                        hipFuncAttributeMaxDynamicSharedMemorySize, SMEM_BYTES);
    dim3 grid(HB_B * LO_TILES, F_TILES);    // (512, 4)
    hyp_conv_wmma_tdm<<<grid, dim3(256), SMEM_BYTES, stream>>>(xhi, xlo, wt, bias, out);
}
